// SpectralCell_19344532701187
// MI455X (gfx1250) — compile-verified, matrix operand order checked by reading, not run on hardware
//
#include <hip/hip_runtime.h>
#include <math.h>

// ---------------------------------------------------------------------------
// SpectralCell forward for MI455X (gfx1250, wave32, WMMA).
// All dense linears run on v_wmma_f32_16x16x32_f16 (f16 in, f32 accum) with
// fused bias / exact-GELU / residual epilogues (bandwidth-bound workload:
// touch each activation tensor once). K is a template parameter so the
// reduction loop fully unrolls into back-to-back v_wmma ops.
// 4x4 eigh = per-row f64 cyclic Jacobi (matches reference float64 section).
// ---------------------------------------------------------------------------

typedef __attribute__((ext_vector_type(16))) _Float16 v16h;
typedef __attribute__((ext_vector_type(8)))  float    v8f;

#define BN   32768   // B*N rows
#define HID  128
#define TOKD 64

__device__ __forceinline__ float gelu_exact(float x) {
  return 0.5f * x * (1.0f + erff(x * 0.70710678118654752f));
}

// ---------------------------------------------------------------------------
// GEMM: O[M x N] = act(A[M x K] @ W[N x K]^T + bias) (+ R[M x N])
// One wave per 16x16 output tile. K in {64,128} (compile-time), N mult of 16.
// Lane layouts follow CDNA5 ISA 7.12.2 (16-bit A 16x32, B 32x16, f32 C/D).
// ---------------------------------------------------------------------------
template<int ACT, bool RES, int K>
__global__ __launch_bounds__(128) void k_gemm_wmma(
    const float* __restrict__ A, const float* __restrict__ W,
    const float* __restrict__ bias, const float* __restrict__ R,
    float* __restrict__ O, int Mrows, int N)
{
  int wave = blockIdx.x * (blockDim.x >> 5) + (threadIdx.x >> 5);
  int lane = threadIdx.x & 31;
  int ntile = N >> 4;
  int rowT = wave / ntile;
  int colT = wave - rowT * ntile;
  int row0 = rowT << 4;
  int n0   = colT << 4;
  if (row0 >= Mrows) return;               // wave-uniform: EXEC stays all-1s
  int half = lane >> 4;                    // K-half selector
  int lm   = lane & 15;                    // M row (A) / N col (B,C,D)

  const float* Ab0 = A + (size_t)(row0 + lm) * K + half * 8;
  const float* Wb0 = W + (size_t)(n0  + lm) * K + half * 8;

  v8f c = {};
#pragma unroll
  for (int k0 = 0; k0 < K; k0 += 32) {
    const float* Ap = Ab0 + k0;
    const float* Wp = Wb0 + k0;
    v16h a = {}, b = {};
#pragma unroll
    for (int e = 0; e < 8; ++e) {
      // lane<16: K = e (elems 0..7), 16+e (elems 8..15); lane>=16: +8
      a[e]     = (_Float16)Ap[e];
      a[e + 8] = (_Float16)Ap[e + 16];
      b[e]     = (_Float16)Wp[e];
      b[e + 8] = (_Float16)Wp[e + 16];
    }
    // (neg_a, A, neg_b, B, c_mod, C, reuse_a, reuse_b)
    c = __builtin_amdgcn_wmma_f32_16x16x32_f16(
        false, a, false, b, (short)0, c, false, false);
  }

  int n = n0 + lm;
  float bn = bias[n];
#pragma unroll
  for (int r = 0; r < 8; ++r) {
    int m = row0 + r + half * 8;           // C/D layout: VGPR r -> M = r (+8)
    float v = c[r] + bn;
    if (ACT == 1) v = gelu_exact(v);
    if (RES) v += R[(size_t)m * N + n];
    O[(size_t)m * N + n] = v;
  }
}

// ---------------------------------------------------------------------------
// LayerNorm over 128 features, one wave per row.
// ---------------------------------------------------------------------------
__global__ __launch_bounds__(256) void k_ln128(
    const float* __restrict__ X, const float* __restrict__ g,
    const float* __restrict__ b, float* __restrict__ O, int Mrows)
{
  int wave = blockIdx.x * (blockDim.x >> 5) + (threadIdx.x >> 5);
  int lane = threadIdx.x & 31;
  if (wave >= Mrows) return;
  const float* x = X + (size_t)wave * HID;
  float v0 = x[lane], v1 = x[lane + 32], v2 = x[lane + 64], v3 = x[lane + 96];
  float s = v0 + v1 + v2 + v3;
  for (int o = 16; o >= 1; o >>= 1) s += __shfl_xor(s, o, 32);
  float mean = s * (1.0f / 128.0f);
  float d0 = v0 - mean, d1 = v1 - mean, d2 = v2 - mean, d3 = v3 - mean;
  float q = d0 * d0 + d1 * d1 + d2 * d2 + d3 * d3;
  for (int o = 16; o >= 1; o >>= 1) q += __shfl_xor(q, o, 32);
  float inv = rsqrtf(q * (1.0f / 128.0f) + 1e-5f);
  float* op = O + (size_t)wave * HID;
  op[lane]      = d0 * inv * g[lane]      + b[lane];
  op[lane + 32] = d1 * inv * g[lane + 32] + b[lane + 32];
  op[lane + 64] = d2 * inv * g[lane + 64] + b[lane + 64];
  op[lane + 96] = d3 * inv * g[lane + 96] + b[lane + 96];
}

// ---------------------------------------------------------------------------
// Row-normalize M (in place), 4x4 f64 Gram + cyclic Jacobi eigh, emit S,U,Vt.
// One thread per row (matches reference's float64 path).
// ---------------------------------------------------------------------------
__global__ __launch_bounds__(256) void k_norm_eig(
    float* __restrict__ Mb, float* __restrict__ U,
    float* __restrict__ Vt, float* __restrict__ S, int Mrows)
{
  int r = blockIdx.x * blockDim.x + threadIdx.x;
  if (r >= Mrows) return;
  float m[16][4];
  float* mp = Mb + (size_t)r * 64;
#pragma unroll
  for (int v = 0; v < 16; ++v) {
    float a0 = mp[v * 4 + 0], a1 = mp[v * 4 + 1];
    float a2 = mp[v * 4 + 2], a3 = mp[v * 4 + 3];
    float nn = sqrtf(a0 * a0 + a1 * a1 + a2 * a2 + a3 * a3);
    float inv = 1.0f / fmaxf(nn, 1e-12f);
    m[v][0] = a0 * inv; m[v][1] = a1 * inv;
    m[v][2] = a2 * inv; m[v][3] = a3 * inv;
    mp[v * 4 + 0] = m[v][0]; mp[v * 4 + 1] = m[v][1];
    mp[v * 4 + 2] = m[v][2]; mp[v * 4 + 3] = m[v][3];
  }
  double G[4][4];
  for (int d = 0; d < 4; ++d)
    for (int e = 0; e < 4; ++e) {
      double s = 0.0;
      for (int v = 0; v < 16; ++v) s += (double)m[v][d] * (double)m[v][e];
      G[d][e] = s;
    }
  double Vv[4][4] = {{1,0,0,0},{0,1,0,0},{0,0,1,0},{0,0,0,1}};
  for (int sweep = 0; sweep < 12; ++sweep) {
    for (int p = 0; p < 3; ++p)
      for (int q = p + 1; q < 4; ++q) {
        double apq = G[p][q];
        if (fabs(apq) < 1e-300) continue;
        double tau = (G[q][q] - G[p][p]) / (2.0 * apq);
        double t = (tau >= 0.0) ?  1.0 / ( tau + sqrt(1.0 + tau * tau))
                                : -1.0 / (-tau + sqrt(1.0 + tau * tau));
        double cth = 1.0 / sqrt(1.0 + t * t);
        double sth = t * cth;
        for (int k = 0; k < 4; ++k) {
          double gpk = G[p][k], gqk = G[q][k];
          G[p][k] = cth * gpk - sth * gqk;
          G[q][k] = sth * gpk + cth * gqk;
        }
        for (int k = 0; k < 4; ++k) {
          double gkp = G[k][p], gkq = G[k][q];
          G[k][p] = cth * gkp - sth * gkq;
          G[k][q] = sth * gkp + cth * gkq;
        }
        for (int k = 0; k < 4; ++k) {
          double vkp = Vv[k][p], vkq = Vv[k][q];
          Vv[k][p] = cth * vkp - sth * vkq;
          Vv[k][q] = sth * vkp + cth * vkq;
        }
      }
  }
  double w[4] = {G[0][0], G[1][1], G[2][2], G[3][3]};
  for (int a = 0; a < 3; ++a) {             // sort descending, swap eigvec cols
    int best = a;
    for (int bb = a + 1; bb < 4; ++bb) if (w[bb] > w[best]) best = bb;
    if (best != a) {
      double tw = w[a]; w[a] = w[best]; w[best] = tw;
      for (int k = 0; k < 4; ++k) {
        double tv = Vv[k][a]; Vv[k][a] = Vv[k][best]; Vv[k][best] = tv;
      }
    }
  }
  float* Sp = S  + (size_t)r * 4;
  float* Up = U  + (size_t)r * 64;
  float* Vp = Vt + (size_t)r * 16;
  float Sf[4];
  for (int d = 0; d < 4; ++d) { Sf[d] = (float)sqrt(fmax(w[d], 0.0)); Sp[d] = Sf[d]; }
  for (int d = 0; d < 4; ++d) {
    double inv = 1.0 / fmax((double)Sf[d], 1e-6);
    for (int e = 0; e < 4; ++e) Vp[d * 4 + e] = (float)Vv[e][d];
    for (int v = 0; v < 16; ++v) {
      double s = 0.0;
      for (int e = 0; e < 4; ++e) s += (double)m[v][e] * Vv[e][d];
      Up[v * 4 + d] = (float)(s * inv);
    }
  }
}

// ---------------------------------------------------------------------------
// Pairwise distances + per-component MLP (15 -> 32 -> 8), one thread per row.
// chunks[k][p] = d2[p*8 + k] (reference reshape/transpose).
// ---------------------------------------------------------------------------
__global__ __launch_bounds__(256) void k_geo(
    const float* __restrict__ Mb, const float* __restrict__ w1,
    const float* __restrict__ b1, const float* __restrict__ w2,
    const float* __restrict__ b2, float* __restrict__ Geo, int Mrows)
{
  int r = blockIdx.x * blockDim.x + threadIdx.x;
  if (r >= Mrows) return;
  float m[16][4], nrm[16];
  const float* mp = Mb + (size_t)r * 64;
#pragma unroll
  for (int v = 0; v < 16; ++v) {
    m[v][0] = mp[v * 4 + 0]; m[v][1] = mp[v * 4 + 1];
    m[v][2] = mp[v * 4 + 2]; m[v][3] = mp[v * 4 + 3];
    nrm[v] = m[v][0] * m[v][0] + m[v][1] * m[v][1]
           + m[v][2] * m[v][2] + m[v][3] * m[v][3];
  }
  float* gp = Geo + (size_t)r * 64;
  for (int k = 0; k < 8; ++k) {
    float pv[15];
    for (int p = 0; p < 15; ++p) {
      int t = p * 8 + k;                    // linear triu index -> (i,j)
      int i = 0, rem = t;
      while (rem >= 15 - i) { rem -= 15 - i; ++i; }
      int j = i + 1 + rem;
      float dot = m[i][0] * m[j][0] + m[i][1] * m[j][1]
                + m[i][2] * m[j][2] + m[i][3] * m[j][3];
      pv[p] = fmaxf(0.0f, nrm[i] + nrm[j] - 2.0f * dot);
    }
    float acc[8];
    for (int o = 0; o < 8; ++o) acc[o] = b2[k * 8 + o];
    for (int h = 0; h < 32; ++h) {
      float s = b1[k * 32 + h];
      for (int p = 0; p < 15; ++p) s += pv[p] * w1[(k * 32 + h) * 15 + p];
      float gl = gelu_exact(s);
      for (int o = 0; o < 8; ++o) acc[o] += gl * w2[(k * 8 + o) * 32 + h];
    }
    for (int o = 0; o < 8; ++o) gp[k * 8 + o] = acc[o];
  }
}

// ---------------------------------------------------------------------------
// Spectral attention pieces (D=4, NHEADS=2, head dim 2).
// ---------------------------------------------------------------------------
__global__ __launch_bounds__(256) void k_qkv(
    const float* __restrict__ Sb, const float* __restrict__ lg,
    const float* __restrict__ lb, const float* __restrict__ qw,
    const float* __restrict__ qb, float* __restrict__ QKV, int Mrows)
{
  int r = blockIdx.x * blockDim.x + threadIdx.x;
  if (r >= Mrows) return;
  const float* sp = Sb + (size_t)r * 4;
  float s0 = sp[0], s1 = sp[1], s2 = sp[2], s3 = sp[3];
  float mean = 0.25f * (s0 + s1 + s2 + s3);
  float d0 = s0 - mean, d1 = s1 - mean, d2 = s2 - mean, d3 = s3 - mean;
  float var = 0.25f * (d0 * d0 + d1 * d1 + d2 * d2 + d3 * d3);
  float inv = rsqrtf(var + 1e-5f);
  float sn[4] = { d0 * inv * lg[0] + lb[0], d1 * inv * lg[1] + lb[1],
                  d2 * inv * lg[2] + lb[2], d3 * inv * lg[3] + lb[3] };
  float* qp = QKV + (size_t)r * 12;
  for (int o = 0; o < 12; ++o)
    qp[o] = qw[o * 4 + 0] * sn[0] + qw[o * 4 + 1] * sn[1]
          + qw[o * 4 + 2] * sn[2] + qw[o * 4 + 3] * sn[3] + qb[o];
}

// One wave per (batch, head, query): streaming softmax over 1024 keys.
__global__ __launch_bounds__(256) void k_attn(
    const float* __restrict__ QKV, float* __restrict__ Ob)
{
  int wave = blockIdx.x * (blockDim.x >> 5) + (threadIdx.x >> 5);
  int lane = threadIdx.x & 31;
  int b = wave >> 11;
  int rem = wave & 2047;
  int h = rem >> 10;
  int n = rem & 1023;
  const float* base = QKV + (size_t)(b * 1024) * 12;
  const float* qp = base + (size_t)n * 12 + h * 2;
  float q0 = qp[0], q1 = qp[1];
  const float scale = 0.70710678118654752f;   // hd^-0.5, hd = 2
  float mx = -3.0e38f, sum = 0.f, a0 = 0.f, a1 = 0.f;
  for (int j = lane; j < 1024; j += 32) {
    const float* kp = base + (size_t)j * 12 + 4 + h * 2;   // K; V at kp[4..5]
    float sc = (q0 * kp[0] + q1 * kp[1]) * scale;
    float nm = fmaxf(mx, sc);
    float corr = __expf(mx - nm);
    float wgt  = __expf(sc - nm);
    sum = sum * corr + wgt;
    a0  = a0  * corr + wgt * kp[4];
    a1  = a1  * corr + wgt * kp[5];
    mx = nm;
  }
  for (int o = 16; o >= 1; o >>= 1) {
    float mx2 = __shfl_xor(mx, o, 32);
    float sm2 = __shfl_xor(sum, o, 32);
    float b0  = __shfl_xor(a0, o, 32);
    float b1  = __shfl_xor(a1, o, 32);
    float nm = fmaxf(mx, mx2);
    float c1 = __expf(mx - nm), c2 = __expf(mx2 - nm);
    sum = sum * c1 + sm2 * c2;
    a0  = a0  * c1 + b0  * c2;
    a1  = a1  * c1 + b1  * c2;
    mx = nm;
  }
  if (lane == 0) {
    float inv = 1.0f / sum;
    float* op = Ob + (size_t)(b * 1024 + n) * 4 + h * 2;
    op[0] = a0 * inv;
    op[1] = a1 * inv;
  }
}

__global__ __launch_bounds__(256) void k_gate(
    float* __restrict__ Sb, const float* __restrict__ Ob,
    const float* __restrict__ ow, const float* __restrict__ obias,
    const float* __restrict__ alog, int Mrows)
{
  int r = blockIdx.x * blockDim.x + threadIdx.x;
  if (r >= Mrows) return;
  const float* op = Ob + (size_t)r * 4;
  float o0 = op[0], o1 = op[1], o2 = op[2], o3 = op[3];
  for (int d = 0; d < 4; ++d) {
    float g = tanhf(ow[d * 4 + 0] * o0 + ow[d * 4 + 1] * o1
                  + ow[d * 4 + 2] * o2 + ow[d * 4 + 3] * o3 + obias[d]);
    float al = 0.2f / (1.0f + __expf(-alog[d]));
    Sb[(size_t)r * 4 + d] *= (1.0f + al * g);
  }
}

// M_hat = U diag(S2) Vt, one thread per row.
__global__ __launch_bounds__(256) void k_mhat(
    const float* __restrict__ U, const float* __restrict__ S,
    const float* __restrict__ Vt, float* __restrict__ Mh, int Mrows)
{
  int r = blockIdx.x * blockDim.x + threadIdx.x;
  if (r >= Mrows) return;
  const float* Up = U  + (size_t)r * 64;
  const float* Sp = S  + (size_t)r * 4;
  const float* Vp = Vt + (size_t)r * 16;
  float sv[4][4];
  for (int d = 0; d < 4; ++d)
    for (int e = 0; e < 4; ++e) sv[d][e] = Sp[d] * Vp[d * 4 + e];
  float* op = Mh + (size_t)r * 64;
  for (int v = 0; v < 16; ++v)
    for (int e = 0; e < 4; ++e) {
      float s = 0.f;
      for (int d = 0; d < 4; ++d) s += Up[v * 4 + d] * sv[d][e];
      op[v * 4 + e] = s;
    }
}

// ---------------------------------------------------------------------------
// Host orchestration.
// ---------------------------------------------------------------------------
extern "C" void kernel_launch(void* const* d_in, const int* in_sizes, int n_in,
                              void* d_out, int out_size, void* d_ws, size_t ws_size,
                              hipStream_t stream) {
  (void)in_sizes; (void)n_in; (void)out_size; (void)ws_size;

  // --- unpack inputs: setup_inputs() dict insertion order (46 leaves) ---
  int ip = 0;
  auto nxt = [&]() { return (const float*)d_in[ip++]; };
  const float* tokens   = nxt();
  const float* enc_in_w = nxt();
  const float* enc_in_b = nxt();
  struct Blk { const float *ln_g, *ln_b, *w1, *b1, *w2, *b2; } eb[2], ob[2];
  for (int i = 0; i < 2; ++i) {
    eb[i].ln_g = nxt(); eb[i].ln_b = nxt(); eb[i].w1 = nxt();
    eb[i].b1 = nxt();   eb[i].w2 = nxt();   eb[i].b2 = nxt();
  }
  const float* enc_out_w = nxt();
  const float* enc_out_b = nxt();
  const float* ca_ln_g = nxt(); const float* ca_ln_b = nxt();
  const float* ca_qkv_w = nxt(); const float* ca_qkv_b = nxt();
  const float* ca_out_w = nxt(); const float* ca_out_b = nxt();
  const float* ca_alpha = nxt();
  const float* pw_w1 = nxt(); const float* pw_b1 = nxt();
  const float* pw_w2 = nxt(); const float* pw_b2 = nxt();
  const float* geo_w = nxt(); const float* geo_b = nxt();
  const float* out_in_w = nxt(); const float* out_in_b = nxt();
  for (int i = 0; i < 2; ++i) {
    ob[i].ln_g = nxt(); ob[i].ln_b = nxt(); ob[i].w1 = nxt();
    ob[i].b1 = nxt();   ob[i].w2 = nxt();   ob[i].b2 = nxt();
  }
  const float* out_proj_w = nxt();
  const float* out_proj_b = nxt();

  // --- workspace layout (floats), regions reused across phases ---
  float* ws   = (float*)d_ws;
  float* Abuf = ws;                        // h / h2          : BN*128
  float* Tbuf = Abuf + (size_t)BN * 128;   // LN scratch; geo : BN*128
  float* Cbuf = Tbuf + (size_t)BN * 128;   // gelu u; qkv; Mhat: BN*128
  float* Mbuf = Cbuf + (size_t)BN * 128;   // normalized M    : BN*64
  float* Umat = Mbuf + (size_t)BN * 64;    // U               : BN*64
  float* Vtm  = Umat + (size_t)BN * 64;    // Vt              : BN*16
  float* Svec = Vtm  + (size_t)BN * 16;    // S / Sb in place : BN*4
  float* Obuf = Cbuf + (size_t)BN * 12;    // attn output o   : BN*4 (inside C)
  float* Geo  = Tbuf;                      // geo features    : BN*64 (inside T)
  float* Mhat = Cbuf;                      // M_hat           : BN*64 (inside C)

  auto gemm_grid = [](int M, int N) { return ((M >> 4) * (N >> 4)) >> 2; };
  const int ROWB = 256, ROWG = BN / 256;     // per-row kernels
  const int LNG = BN / 8;                    // LN: 8 waves/block

  // Phase 1: encoder
  k_gemm_wmma<0,false,TOKD><<<gemm_grid(BN,HID), 128, 0, stream>>>(
      tokens, enc_in_w, enc_in_b, nullptr, Abuf, BN, HID);
  for (int i = 0; i < 2; ++i) {
    k_ln128<<<LNG, 256, 0, stream>>>(Abuf, eb[i].ln_g, eb[i].ln_b, Tbuf, BN);
    k_gemm_wmma<1,false,HID><<<gemm_grid(BN,HID), 128, 0, stream>>>(
        Tbuf, eb[i].w1, eb[i].b1, nullptr, Cbuf, BN, HID);
    k_gemm_wmma<0,true,HID><<<gemm_grid(BN,HID), 128, 0, stream>>>(
        Cbuf, eb[i].w2, eb[i].b2, Abuf, Abuf, BN, HID);
  }
  k_gemm_wmma<0,false,HID><<<gemm_grid(BN,64), 128, 0, stream>>>(
      Abuf, enc_out_w, enc_out_b, nullptr, Mbuf, BN, 64);

  // Phase 2: normalize + eigh, geometry features
  k_norm_eig<<<ROWG, ROWB, 0, stream>>>(Mbuf, Umat, Vtm, Svec, BN);
  k_geo<<<ROWG, ROWB, 0, stream>>>(Mbuf, pw_w1, pw_b1, pw_w2, pw_b2, Geo, BN);

  // Phase 3: spectral cross-attention (NCROSS = 1), Sb updated in place
  k_qkv<<<ROWG, ROWB, 0, stream>>>(Svec, ca_ln_g, ca_ln_b, ca_qkv_w, ca_qkv_b,
                                   Cbuf, BN);
  k_attn<<<(32 * 2 * 1024) / 8, 256, 0, stream>>>(Cbuf, Obuf);
  k_gate<<<ROWG, ROWB, 0, stream>>>(Svec, Obuf, ca_out_w, ca_out_b, ca_alpha, BN);

  // Phase 4: reconstruct M_hat
  k_mhat<<<ROWG, ROWB, 0, stream>>>(Umat, Svec, Vtm, Mhat, BN);

  // Phase 5: h2 = M_hat @ out_in^T + b  +  geo @ geo_w^T + geo_b
  k_gemm_wmma<0,false,TOKD><<<gemm_grid(BN,HID), 128, 0, stream>>>(
      Mhat, out_in_w, out_in_b, nullptr, Abuf, BN, HID);
  k_gemm_wmma<0,true,TOKD><<<gemm_grid(BN,HID), 128, 0, stream>>>(
      Geo, geo_w, geo_b, Abuf, Abuf, BN, HID);

  // Phase 6: output blocks
  for (int i = 0; i < 2; ++i) {
    k_ln128<<<LNG, 256, 0, stream>>>(Abuf, ob[i].ln_g, ob[i].ln_b, Tbuf, BN);
    k_gemm_wmma<1,false,HID><<<gemm_grid(BN,HID), 128, 0, stream>>>(
        Tbuf, ob[i].w1, ob[i].b1, nullptr, Cbuf, BN, HID);
    k_gemm_wmma<0,true,HID><<<gemm_grid(BN,HID), 128, 0, stream>>>(
        Cbuf, ob[i].w2, ob[i].b2, Abuf, Abuf, BN, HID);
  }

  // Phase 7: output projection straight into d_out
  k_gemm_wmma<0,false,HID><<<gemm_grid(BN,TOKD), 128, 0, stream>>>(
      Abuf, out_proj_w, out_proj_b, nullptr, (float*)d_out, BN, TOKD);
}
